// CausalGroupedQueryAttention_71425306132772
// MI455X (gfx1250) — compile-verified, matrix operand order checked by reading, not run on hardware
//
#include <hip/hip_runtime.h>
#include <hip/hip_bf16.h>

// ---------------------------------------------------------------------------
// CDNA5 (gfx1250) GQA attention block: QKV proj + RMSNorm + RoPE +
// causal softcap flash attention + output proj, all matmuls via
// v_wmma_f32_16x16x32_bf16 (wave32). GEMM A-tiles staged with the
// Tensor Data Mover (tensor_load_to_lds + s_wait_tensorcnt).
// ---------------------------------------------------------------------------

typedef __bf16 bf16_t;
typedef __attribute__((ext_vector_type(16))) __bf16 v16bf;
typedef __attribute__((ext_vector_type(8)))  __bf16 v8bf;
typedef __attribute__((ext_vector_type(8)))  float  v8f;
typedef __attribute__((ext_vector_type(4)))  float  f32x4;
typedef __attribute__((ext_vector_type(4)))  unsigned int u32x4;
typedef __attribute__((ext_vector_type(4)))  unsigned int uint32x4_t;
typedef __attribute__((ext_vector_type(8)))  int int32x8_t;
typedef __attribute__((ext_vector_type(4)))  int int32x4_t;

#define WMMA_BF16(A, B, C) \
  __builtin_amdgcn_wmma_f32_16x16x32_bf16(false, (A), false, (B), (short)0, (C), false, false)

#if __has_builtin(__builtin_amdgcn_tensor_load_to_lds) && \
    __has_builtin(__builtin_amdgcn_s_wait_tensorcnt)
#define USE_TDM 1
#else
#define USE_TDM 0
#endif

union FragU { v16bf v; u32x4 u[2]; };

static constexpr int   T_SEQ    = 4096;
static constexpr int   C_DIM    = 2048;
static constexpr int   NH       = 16;
static constexpr int   NKV      = 4;
static constexpr int   HD       = 128;
static constexpr float SOFTCAP  = 50.0f;
static constexpr float INV_SQRT_HD = 0.08838834764831845f; // 1/sqrt(128)

// ---------------------------------------------------------------------------
// Kernel 0: fp32 -> bf16 conversion, 8 elements / thread (2x b128 load, b128 store)
// ---------------------------------------------------------------------------
__global__ void cvt_f32_bf16(const float* __restrict__ in, bf16_t* __restrict__ out, int n8) {
  const int i = blockIdx.x * blockDim.x + threadIdx.x;
  if (i < n8) {
    const f32x4 a = ((const f32x4*)in)[2 * i];
    const f32x4 b = ((const f32x4*)in)[2 * i + 1];
    v8bf o;
    o[0] = (bf16_t)a[0]; o[1] = (bf16_t)a[1]; o[2] = (bf16_t)a[2]; o[3] = (bf16_t)a[3];
    o[4] = (bf16_t)b[0]; o[5] = (bf16_t)b[1]; o[6] = (bf16_t)b[2]; o[7] = (bf16_t)b[3];
    ((v8bf*)out)[i] = o;
  }
}

// ---------------------------------------------------------------------------
// Kernel 1: bf16 GEMM  C(f32, MxN) = A(bf16, MxK, rowmajor) @ B(bf16, KxN, rowmajor)
// N, K compile-time so all strides fold into immediate offsets.
// Block tile 128x128, 8 waves (4x2), K-step 32.
// A tile (128 rows x 64B, contiguous rows) is staged by the Tensor Data Mover:
// wave 0 issues one tensor_load_to_lds per K-step (D# built per ISA 8.3/8.4),
// waits on TENSORcnt, then the workgroup barrier publishes the LDS tile.
// Fragment packing (ISA 7.12.2, 16-bit A 16x32):
//   lanes 0-15 : K {0..7, 16..23} ; lanes 16-31 : K {8..15, 24..31}
// B mirrors A with N replacing M (lane = output column).
// ---------------------------------------------------------------------------
template <int N, int K>
__global__ __launch_bounds__(256) void gemm_bf16(const bf16_t* __restrict__ A,
                                                 const bf16_t* __restrict__ B,
                                                 float* __restrict__ C) {
  __shared__ bf16_t As[128 * 32];   // [m][k]
  __shared__ bf16_t Bs[128 * 32];   // [n][k] (transposed from global)

  const int tid  = threadIdx.x;
  const int wave = tid >> 5;
  const int lane = tid & 31;
  const int l16  = lane & 15;
  const int half = lane >> 4;
  const int mw   = wave & 3;   // 4 waves along M -> 32 rows each
  const int nw   = wave >> 2;  // 2 waves along N -> 64 cols each
  const int blockM = blockIdx.y * 128;
  const int blockN = blockIdx.x * 128;

  const v8f vzero = {0.f,0.f,0.f,0.f,0.f,0.f,0.f,0.f};
  v8f acc[2][4];
  #pragma unroll
  for (int mi = 0; mi < 2; ++mi)
    #pragma unroll
    for (int ni = 0; ni < 4; ++ni) acc[mi][ni] = vzero;

  // per-thread staging pointers (advance by compile-time increments)
  const bf16_t* bSrc = B + (size_t)(tid >> 3) * N + blockN + (tid & 7) * 16;
  bf16_t* bDst = Bs + ((tid & 7) * 16) * 32 + (tid >> 3);

#if USE_TDM
  const unsigned long long gaBase =
      (unsigned long long)(uintptr_t)(A + (size_t)blockM * K);
  const unsigned ldsAs = (unsigned)(uintptr_t)(&As[0]);
  // D# group1: data_size=4B (code 2); tile 16 dwords x 128 rows;
  // tensor dims = tile dims (no OOB); row stride = K/2 dwords.
  const int32x8_t g1 = { (int)0x20000u,          // data_size=2 (4B)
                         (int)(16u << 16),       // tensor_dim0[15:0]=16
                         (int)(128u << 16),      // tensor_dim1[15:0]=128
                         (int)(16u << 16),       // tile_dim0=16
                         (int)128,               // tile_dim1=128, tile_dim2=0
                         (int)(K / 2),           // tensor_dim0_stride low32
                         0, 0 };
  const int32x4_t gz = { 0, 0, 0, 0 };
#else
  const bf16_t* aSrc = A + (size_t)(blockM + (tid >> 1)) * K + (tid & 1) * 16;
  u32x4* aDst = (u32x4*)(As + (tid >> 1) * 32 + (tid & 1) * 16);
#endif

  for (int k0 = 0; k0 < K; k0 += 32) {
    __syncthreads();
#if USE_TDM
    // wave 0 issues the TDM copy of the A tile into LDS
    if (wave == 0) {
      const unsigned long long ga = gaBase + (unsigned long long)k0 * 2ull;
      const uint32x4_t g0 = { 1u,                                   // count=1
                              ldsAs,                                // lds_addr
                              (unsigned)ga,                         // gaddr[31:0]
                              ((unsigned)(ga >> 32) & 0x1FFFFFFu) | // gaddr[56:32]
                                  0x80000000u };                    // type=2
#if defined(__clang_major__) && __clang_major__ >= 23
      const int32x8_t gz8 = { 0, 0, 0, 0, 0, 0, 0, 0 };
      __builtin_amdgcn_tensor_load_to_lds(g0, g1, gz, gz, gz8, 0);
#else
      __builtin_amdgcn_tensor_load_to_lds(g0, g1, gz, gz, 0);
#endif
    }
#else
    {
      const u32x4* s = (const u32x4*)aSrc;
      aDst[0] = s[0];
      aDst[1] = s[1];
      aSrc += 32;
    }
#endif
    // stage B tile transposed: 16 contiguous n per thread -> Bs[n][k]
    {
      const u32x4* s = (const u32x4*)bSrc;
      const u32x4 b0 = s[0];
      const u32x4 b1 = s[1];
      const bf16_t* e0 = (const bf16_t*)&b0;
      const bf16_t* e1 = (const bf16_t*)&b1;
      #pragma unroll
      for (int i = 0; i < 8; ++i) bDst[i * 32] = e0[i];
      #pragma unroll
      for (int i = 0; i < 8; ++i) bDst[(i + 8) * 32] = e1[i];
      bSrc += (size_t)32 * N;
    }
    // prefetch next B tile while WMMAs run
    if (k0 + 32 < K) __builtin_prefetch(bSrc, 0, 1);
#if USE_TDM
    if (wave == 0) __builtin_amdgcn_s_wait_tensorcnt(0);
#endif
    __syncthreads();

    v16bf afr[2], bfr[4];
    #pragma unroll
    for (int mi = 0; mi < 2; ++mi) {
      const int row = mw * 32 + mi * 16 + l16;
      const u32x4* ch = (const u32x4*)(As + row * 32);
      FragU f; f.u[0] = ch[half]; f.u[1] = ch[half + 2];
      afr[mi] = f.v;
    }
    #pragma unroll
    for (int ni = 0; ni < 4; ++ni) {
      const int col = nw * 64 + ni * 16 + l16;
      const u32x4* ch = (const u32x4*)(Bs + col * 32);
      FragU f; f.u[0] = ch[half]; f.u[1] = ch[half + 2];
      bfr[ni] = f.v;
    }
    #pragma unroll
    for (int mi = 0; mi < 2; ++mi)
      #pragma unroll
      for (int ni = 0; ni < 4; ++ni)
        acc[mi][ni] = WMMA_BF16(afr[mi], bfr[ni], acc[mi][ni]);
  }

  // epilogue: C layout -> lane = col, VGPR r = row (r + 8*half).
  // Compile-time N => all 64 stores are base + immediate offset.
  float* cp = C + (size_t)(blockM + mw * 32 + half * 8) * N
                + blockN + nw * 64 + l16;
  #pragma unroll
  for (int mi = 0; mi < 2; ++mi)
    #pragma unroll
    for (int ni = 0; ni < 4; ++ni)
      #pragma unroll
      for (int r = 0; r < 8; ++r)
        cp[(size_t)(mi * 16 + r) * N + ni * 16] = acc[mi][ni][r];
}

// ---------------------------------------------------------------------------
// Kernel 2: per-token, per-head RMSNorm + RoPE; reshape to head-major bf16.
// 128 threads = 4 waves; each wave owns one head at a time (32 lanes x 4 dims).
// Lane holds dims {lane, lane+32, lane+64, lane+96} -> rope pairs (d, d+64).
// ---------------------------------------------------------------------------
__global__ __launch_bounds__(128) void qk_norm_rope(
    const float* __restrict__ Qf,   // [T][NH*HD]
    const float* __restrict__ KVf,  // [T][2*NKV*HD]
    const float* __restrict__ qn_w, const float* __restrict__ kn_w,
    bf16_t* __restrict__ Qb,        // [NH][T][HD]
    bf16_t* __restrict__ Kb,        // [NKV][T][HD]
    bf16_t* __restrict__ Vb) {      // [NKV][T][HD]
  const int t    = blockIdx.x;
  const int wave = threadIdx.x >> 5;
  const int lane = threadIdx.x & 31;
  const float tf = (float)t;

  // rope angles for pair indices lane and lane+32
  float a0 = tf * __powf(10000.0f, -(float)lane * (1.0f / 64.0f));
  float a1 = tf * __powf(10000.0f, -(float)(lane + 32) * (1.0f / 64.0f));
  float c0, s0, c1, s1;
  __sincosf(a0, &s0, &c0);
  __sincosf(a1, &s1, &c1);

  // Q heads
  for (int h = wave; h < NH; h += 4) {
    const float* src = Qf + (size_t)t * (NH * HD) + h * HD;
    float e0 = src[lane], e1 = src[lane + 32], e2 = src[lane + 64], e3 = src[lane + 96];
    float ss = e0 * e0 + e1 * e1 + e2 * e2 + e3 * e3;
    #pragma unroll
    for (int off = 16; off; off >>= 1) ss += __shfl_xor(ss, off, 32);
    const float sc = rsqrtf(ss * (1.0f / HD) + 1e-6f);
    e0 *= sc * qn_w[lane];      e1 *= sc * qn_w[lane + 32];
    e2 *= sc * qn_w[lane + 64]; e3 *= sc * qn_w[lane + 96];
    const float y0 = e0 * c0 + e2 * s0, y2 = -e0 * s0 + e2 * c0;
    const float y1 = e1 * c1 + e3 * s1, y3 = -e1 * s1 + e3 * c1;
    bf16_t* dst = Qb + ((size_t)h * T_SEQ + t) * HD;
    dst[lane] = (bf16_t)y0; dst[lane + 32] = (bf16_t)y1;
    dst[lane + 64] = (bf16_t)y2; dst[lane + 96] = (bf16_t)y3;
  }
  // K head (one per wave) + V copy
  {
    const int h = wave;
    const float* src = KVf + (size_t)t * (2 * NKV * HD) + h * HD;
    float e0 = src[lane], e1 = src[lane + 32], e2 = src[lane + 64], e3 = src[lane + 96];
    float ss = e0 * e0 + e1 * e1 + e2 * e2 + e3 * e3;
    #pragma unroll
    for (int off = 16; off; off >>= 1) ss += __shfl_xor(ss, off, 32);
    const float sc = rsqrtf(ss * (1.0f / HD) + 1e-6f);
    e0 *= sc * kn_w[lane];      e1 *= sc * kn_w[lane + 32];
    e2 *= sc * kn_w[lane + 64]; e3 *= sc * kn_w[lane + 96];
    const float y0 = e0 * c0 + e2 * s0, y2 = -e0 * s0 + e2 * c0;
    const float y1 = e1 * c1 + e3 * s1, y3 = -e1 * s1 + e3 * c1;
    bf16_t* dst = Kb + ((size_t)h * T_SEQ + t) * HD;
    dst[lane] = (bf16_t)y0; dst[lane + 32] = (bf16_t)y1;
    dst[lane + 64] = (bf16_t)y2; dst[lane + 96] = (bf16_t)y3;

    const float* vs = KVf + (size_t)t * (2 * NKV * HD) + NKV * HD + h * HD;
    bf16_t* vd = Vb + ((size_t)h * T_SEQ + t) * HD;
    vd[lane] = (bf16_t)vs[lane];           vd[lane + 32] = (bf16_t)vs[lane + 32];
    vd[lane + 64] = (bf16_t)vs[lane + 64]; vd[lane + 96] = (bf16_t)vs[lane + 96];
  }
}

// ---------------------------------------------------------------------------
// Kernel 3: flash attention (causal, softcap) with WMMA.
// Grid: (T/128, NH). Block 256 thr = 8 waves; wave owns 16 Q rows.
// BLOCK_N = 32 KV tokens per iteration.
// ---------------------------------------------------------------------------
__global__ __launch_bounds__(256) void fa_wmma(
    const bf16_t* __restrict__ Qb,  // [NH][T][HD]
    const bf16_t* __restrict__ Kb,  // [NKV][T][HD]
    const bf16_t* __restrict__ Vb,  // [NKV][T][HD]
    bf16_t* __restrict__ Yb) {      // [T][NH*HD]
  const int h    = blockIdx.y;
  const int kvh  = h >> 2;                  // repeat(k, 4) -> h/4
  const int qbase = blockIdx.x * 128;
  const int wave = threadIdx.x >> 5;
  const int lane = threadIdx.x & 31;
  const int l16  = lane & 15;
  const int half = lane >> 4;
  const int qrow0 = qbase + wave * 16;

  __shared__ bf16_t Vt[HD * 32];            // V block transposed: [d][tok]
  __shared__ bf16_t Ps[8][16 * 32];         // per-wave P transpose scratch [m][c]

  // resident Q fragments (16 rows x 128 dims = 4 A-tiles)
  v16bf qfr[4];
  {
    const bf16_t* qp = Qb + ((size_t)h * T_SEQ + qrow0 + l16) * HD;
    #pragma unroll
    for (int kc = 0; kc < 4; ++kc) {
      const u32x4* ch = (const u32x4*)(qp + kc * 32);
      FragU f; f.u[0] = ch[half]; f.u[1] = ch[half + 2];
      qfr[kc] = f.v;
    }
  }

  const v8f vzero = {0.f,0.f,0.f,0.f,0.f,0.f,0.f,0.f};
  v8f out[8];
  #pragma unroll
  for (int dt = 0; dt < 8; ++dt) out[dt] = vzero;
  float m_run[8], l_run[8];
  #pragma unroll
  for (int r = 0; r < 8; ++r) { m_run[r] = -INFINITY; l_run[r] = 0.0f; }

  const int nblocks = (qbase + 128) / 32;   // causal: only past/diagonal blocks
  for (int j = 0; j < nblocks; ++j) {
    // stage V block transposed into LDS (all waves iterate identically)
    __syncthreads();
    {
      const int tk = threadIdx.x >> 3;         // 0..31
      const int d0 = (threadIdx.x & 7) * 16;   // 0..112
      const bf16_t* vs = Vb + ((size_t)kvh * T_SEQ + j * 32 + tk) * HD + d0;
      #pragma unroll
      for (int i = 0; i < 16; ++i) Vt[(d0 + i) * 32 + tk] = vs[i];
    }
    __syncthreads();

    // scores S = Q @ K^T : 2 tiles of 16 tokens; B frags straight from global
    v8f sacc[2]; sacc[0] = vzero; sacc[1] = vzero;
    #pragma unroll
    for (int nt = 0; nt < 2; ++nt) {
      const bf16_t* kp = Kb + ((size_t)kvh * T_SEQ + j * 32 + nt * 16 + l16) * HD;
      #pragma unroll
      for (int kc = 0; kc < 4; ++kc) {
        const u32x4* ch = (const u32x4*)(kp + kc * 32);
        FragU f; f.u[0] = ch[half]; f.u[1] = ch[half + 2];
        sacc[nt] = WMMA_BF16(qfr[kc], f.v, sacc[nt]);
      }
    }

    // softcap + causal mask + online softmax (per-row stats via 16-lane butterfly)
    float pvals[2][8];
    #pragma unroll
    for (int r = 0; r < 8; ++r) {
      const int row = qrow0 + r + 8 * half;
      float mx = -INFINITY;
      #pragma unroll
      for (int nt = 0; nt < 2; ++nt) {
        const int col = j * 32 + nt * 16 + l16;
        float s = sacc[nt][r] * INV_SQRT_HD;
        s = SOFTCAP * tanhf(s * (1.0f / SOFTCAP));
        s = (col <= row) ? s : -INFINITY;
        pvals[nt][r] = s;
        mx = fmaxf(mx, s);
      }
      #pragma unroll
      for (int off = 8; off; off >>= 1) mx = fmaxf(mx, __shfl_xor(mx, off, 32));
      const float mnew = fmaxf(m_run[r], mx);
      const float alpha = __expf(m_run[r] - mnew);   // exp(-inf - finite) = 0 ok
      float rowsum = 0.0f;
      #pragma unroll
      for (int nt = 0; nt < 2; ++nt) {
        const float p = __expf(pvals[nt][r] - mnew);
        pvals[nt][r] = p;
        rowsum += p;
      }
      #pragma unroll
      for (int off = 8; off; off >>= 1) rowsum += __shfl_xor(rowsum, off, 32);
      l_run[r] = l_run[r] * alpha + rowsum;
      m_run[r] = mnew;
      #pragma unroll
      for (int dt = 0; dt < 8; ++dt) out[dt][r] *= alpha;
    }

    // transpose P (C layout -> A layout) through per-wave LDS scratch;
    // same-wave DS ops are in-order, no barrier needed.
    bf16_t* ps = &Ps[wave][0];
    #pragma unroll
    for (int nt = 0; nt < 2; ++nt)
      #pragma unroll
      for (int r = 0; r < 8; ++r)
        ps[(r + 8 * half) * 32 + nt * 16 + l16] = (bf16_t)pvals[nt][r];

    FragU pf;
    {
      const u32x4* ch = (const u32x4*)(ps + l16 * 32);
      pf.u[0] = ch[half]; pf.u[1] = ch[half + 2];
    }

    // out += P @ V  (8 head-dim tiles, K=32 in one WMMA each)
    #pragma unroll
    for (int dt = 0; dt < 8; ++dt) {
      const u32x4* ch = (const u32x4*)(Vt + (dt * 16 + l16) * 32);
      FragU vf; vf.u[0] = ch[half]; vf.u[1] = ch[half + 2];
      out[dt] = WMMA_BF16(pf.v, vf.v, out[dt]);
    }
  }

  // epilogue: normalize and write token-major bf16 y (constant strides fold)
  bf16_t* yp = Yb + (size_t)(qrow0 + 8 * half) * (NH * HD) + h * HD + l16;
  #pragma unroll
  for (int dt = 0; dt < 8; ++dt)
    #pragma unroll
    for (int r = 0; r < 8; ++r) {
      const float y = out[dt][r] / l_run[r];
      yp[(size_t)r * (NH * HD) + dt * 16] = (bf16_t)y;
    }
}

// ---------------------------------------------------------------------------
// Host-side launcher
// ---------------------------------------------------------------------------
extern "C" void kernel_launch(void* const* d_in, const int* in_sizes, int n_in,
                              void* d_out, int out_size, void* d_ws, size_t ws_size,
                              hipStream_t stream) {
  (void)in_sizes; (void)n_in; (void)out_size; (void)ws_size;

  const float* x    = (const float*)d_in[0];  // [T][C]
  const float* Wq   = (const float*)d_in[1];  // [C][C]
  const float* Wkv  = (const float*)d_in[2];  // [C][2*NKV*HD]
  const float* Wc   = (const float*)d_in[3];  // [C][C]
  const float* qn_w = (const float*)d_in[4];
  const float* kn_w = (const float*)d_in[5];
  float* out = (float*)d_out;

  const size_t MB = 1024ull * 1024ull;
  char* ws = (char*)d_ws;
  bf16_t* xb   = (bf16_t*)(ws + 0);          // 16 MB : x bf16
  bf16_t* Wqb  = (bf16_t*)(ws + 16 * MB);    //  8 MB
  bf16_t* Wkvb = (bf16_t*)(ws + 24 * MB);    //  4 MB
  bf16_t* Wcb  = (bf16_t*)(ws + 28 * MB);    //  8 MB
  float*  Qf   = (float*) (ws + 36 * MB);    // 32 MB : q projection fp32
  float*  KVf  = (float*) (ws + 68 * MB);    // 16 MB : kv projection fp32
  bf16_t* Qb   = (bf16_t*)(ws + 84 * MB);    // 16 MB : [NH][T][HD]
  bf16_t* Kb   = (bf16_t*)(ws + 100 * MB);   //  4 MB : [NKV][T][HD]
  bf16_t* Vb   = (bf16_t*)(ws + 104 * MB);   //  4 MB
  bf16_t* Yb   = (bf16_t*)(ws + 108 * MB);   // 16 MB : [T][C]

  // 0) convert inputs to bf16 (8 elements per thread)
  {
    int n8;
    n8 = (T_SEQ * C_DIM) / 8;
    cvt_f32_bf16<<<(n8 + 255) / 256, 256, 0, stream>>>(x, xb, n8);
    n8 = (C_DIM * C_DIM) / 8;
    cvt_f32_bf16<<<(n8 + 255) / 256, 256, 0, stream>>>(Wq, Wqb, n8);
    n8 = (C_DIM * 2 * NKV * HD) / 8;
    cvt_f32_bf16<<<(n8 + 255) / 256, 256, 0, stream>>>(Wkv, Wkvb, n8);
    n8 = (C_DIM * C_DIM) / 8;
    cvt_f32_bf16<<<(n8 + 255) / 256, 256, 0, stream>>>(Wc, Wcb, n8);
  }

  // 1) projections
  gemm_bf16<C_DIM, C_DIM><<<dim3(C_DIM / 128, T_SEQ / 128), 256, 0, stream>>>(xb, Wqb, Qf);
  gemm_bf16<2 * NKV * HD, C_DIM>
      <<<dim3((2 * NKV * HD) / 128, T_SEQ / 128), 256, 0, stream>>>(xb, Wkvb, KVf);

  // 2) rmsnorm + rope + head-major reshape
  qk_norm_rope<<<T_SEQ, 128, 0, stream>>>(Qf, KVf, qn_w, kn_w, Qb, Kb, Vb);

  // 3) attention
  fa_wmma<<<dim3(T_SEQ / 128, NH), 256, 0, stream>>>(Qb, Kb, Vb, Yb);

  // 4) output projection -> fp32 d_out
  gemm_bf16<C_DIM, C_DIM><<<dim3(C_DIM / 128, T_SEQ / 128), 256, 0, stream>>>(Yb, Wcb, out);
}
